// MagConv_59674275611201
// MI455X (gfx1250) — matrix operand order repaired and checked by reading, not compiled, not run on hardware
//
#include <hip/hip_runtime.h>

typedef __attribute__((ext_vector_type(16))) __bf16 v16bf;
typedef __attribute__((ext_vector_type(8)))  __bf16 v8bf;
typedef __attribute__((ext_vector_type(8)))  float  v8f;

#define K1 3
#define C  128           // IN_C == OUT_C
#define LDS_K_STRIDE 136 // padded contraction-dim stride (bf16 elems) -> bank-conflict-free

// ---------------- kernel 1: zero the f32 accumulators in workspace ----------------
__global__ void MagConv_zero_ws(float* __restrict__ p, long n) {
    long i = (long)blockIdx.x * blockDim.x + threadIdx.x;
    if (i < n) p[i] = 0.0f;
}

// ---------------- kernel 2: edge scatter (SPMM via f32 atomics, L2-resident) ------
// One 32-lane wave group per edge; each lane owns 4 channels (float4).
// Fuses the complex combination: accR[k] += Lr*xr - Li*xi ; accI[k] += Li*xr + Lr*xi
__global__ void MagConv_scatter(const float* __restrict__ Xr,
                                const float* __restrict__ Xi,
                                const float* __restrict__ Lr,
                                const float* __restrict__ Li,
                                const long long* __restrict__ ei, // [2][E] int64
                                float* __restrict__ accR,         // [K1][N][C]
                                float* __restrict__ accI,         // [K1][N][C]
                                int N, int E) {
    long t = (long)blockIdx.x * blockDim.x + threadIdx.x;
    int e = (int)(t >> 5);
    if (e >= E) return;
    int lane = (int)(t & 31);
    e = __builtin_amdgcn_readfirstlane(e);          // provably wave-uniform -> s_load path

    int row = (int)ei[e];                           // edge_index[0] = rows (segment ids)
    int col = (int)ei[(long)E + e];                 // edge_index[1] = cols (gather ids)

    const float4 xr = *(const float4*)(Xr + (long)col * C + lane * 4);
    const float4 xi = *(const float4*)(Xi + (long)col * C + lane * 4);

#pragma unroll
    for (int k = 0; k < K1; ++k) {
        float lr = Lr[(long)k * E + e];
        float li = Li[(long)k * E + e];
        float4 ar, ai;
        ar.x = lr * xr.x - li * xi.x;  ai.x = li * xr.x + lr * xi.x;
        ar.y = lr * xr.y - li * xi.y;  ai.y = li * xr.y + lr * xi.y;
        ar.z = lr * xr.z - li * xi.z;  ai.z = li * xr.z + lr * xi.z;
        ar.w = lr * xr.w - li * xi.w;  ai.w = li * xr.w + lr * xi.w;

        float* pr = accR + ((long)k * N + row) * C + lane * 4;
        float* pi = accI + ((long)k * N + row) * C + lane * 4;
        unsafeAtomicAdd(pr + 0, ar.x);  unsafeAtomicAdd(pr + 1, ar.y);
        unsafeAtomicAdd(pr + 2, ar.z);  unsafeAtomicAdd(pr + 3, ar.w);
        unsafeAtomicAdd(pi + 0, ai.x);  unsafeAtomicAdd(pi + 1, ai.y);
        unsafeAtomicAdd(pi + 2, ai.z);  unsafeAtomicAdd(pi + 3, ai.w);
    }
}

// Load 16 consecutive f32 (as 2x blocks of 8 at p and p+16) and convert to bf16 A-frag.
// A-matrix 16x32 bf16 layout (ISA 7.12.2): lane<16 holds K = {base+0..7, base+16..23},
// lane>=16 holds K = {base+8..15, base+24..31}; 'p' is pre-offset by half*8.
__device__ __forceinline__ v16bf load_cvt_afrag(const float* __restrict__ p) {
    float4 r0 = *(const float4*)(p + 0);
    float4 r1 = *(const float4*)(p + 4);
    float4 r2 = *(const float4*)(p + 16);
    float4 r3 = *(const float4*)(p + 20);
    v16bf a;
    a[0]  = (__bf16)r0.x; a[1]  = (__bf16)r0.y; a[2]  = (__bf16)r0.z; a[3]  = (__bf16)r0.w;
    a[4]  = (__bf16)r1.x; a[5]  = (__bf16)r1.y; a[6]  = (__bf16)r1.z; a[7]  = (__bf16)r1.w;
    a[8]  = (__bf16)r2.x; a[9]  = (__bf16)r2.y; a[10] = (__bf16)r2.z; a[11] = (__bf16)r2.w;
    a[12] = (__bf16)r3.x; a[13] = (__bf16)r3.y; a[14] = (__bf16)r3.z; a[15] = (__bf16)r3.w;
    return a;
}

// ---------------- kernel 3: fused dual GEMM tail with bf16 WMMA -------------------
// out_real = sum_k AccR[k] @ W[k] + bias ; out_imag = sum_k AccI[k] @ W[k] + bias
// 8 waves/block, each wave computes a 16x128 strip of BOTH outputs (16 v8f accums).
__global__ __launch_bounds__(256)
void MagConv_wmma_gemm(const float* __restrict__ accR,
                       const float* __restrict__ accI,
                       const float* __restrict__ W,     // [K1][C][C] (kdim-major rows)
                       const float* __restrict__ bias,  // [C]
                       float* __restrict__ outR,
                       float* __restrict__ outI,
                       int N) {
    // Transposed weights for one order k: lds_wt[n * LDS_K_STRIDE + kd] = W[k][kd][n]
    __shared__ __bf16 lds_wt[C * LDS_K_STRIDE];        // ~34 KB

    const int wave = threadIdx.x >> 5;
    const int lane = threadIdx.x & 31;
    const int half = lane >> 4;
    const int ln   = lane & 15;
    const int row0 = (blockIdx.x * 8 + wave) * 16;
    const bool active = (row0 < N);                    // wave-uniform

    v8f cR[8], cI[8];
#pragma unroll
    for (int j = 0; j < 8; ++j) { cR[j] = 0.0f; cI[j] = 0.0f; }

    for (int k = 0; k < K1; ++k) {
        if (k) __syncthreads();                        // previous k's LDS reads done
        // cooperative stage of W[k] into LDS: coalesced global reads, transposed write
        for (int idx = threadIdx.x; idx < C * C; idx += 256) {
            int kd = idx >> 7;        // contraction dim
            int n  = idx & (C - 1);   // output column
            lds_wt[n * LDS_K_STRIDE + kd] = (__bf16)W[(long)k * C * C + idx];
        }
        __syncthreads();

        if (active) {
            const long rowbase = ((long)k * N + row0 + ln) * C + half * 8;
#pragma unroll
            for (int kc = 0; kc < 4; ++kc) {
                v16bf aR = load_cvt_afrag(accR + rowbase + kc * 32);
                v16bf aI = load_cvt_afrag(accI + rowbase + kc * 32);
#pragma unroll
                for (int j = 0; j < 8; ++j) {
                    // B-frag: 32x16 bf16, lane<16 -> K {0..7,16..23} of column ln, lane>=16 -> K {8..15,24..31}
                    const __bf16* bp = &lds_wt[(j * 16 + ln) * LDS_K_STRIDE + kc * 32 + half * 8];
                    v8bf b0 = *(const v8bf*)(bp);
                    v8bf b1 = *(const v8bf*)(bp + 16);
                    v16bf b;
#pragma unroll
                    for (int q = 0; q < 8; ++q) { b[q] = b0[q]; b[q + 8] = b1[q]; }
                    cR[j] = __builtin_amdgcn_wmma_f32_16x16x32_bf16(
                        false, aR, false, b, (short)0, cR[j], false, false);
                    cI[j] = __builtin_amdgcn_wmma_f32_16x16x32_bf16(
                        false, aI, false, b, (short)0, cI[j], false, false);
                }
            }
        }
    }

    if (active) {
        // C/D layout: VGPR v -> M = v + half*8, N = ln
#pragma unroll
        for (int j = 0; j < 8; ++j) {
            const int colo = j * 16 + ln;
            const float bv = bias[colo];
#pragma unroll
            for (int v = 0; v < 8; ++v) {
                int row = row0 + v + half * 8;
                if (row < N) {
                    outR[(long)row * C + colo] = cR[j][v] + bv;
                    outI[(long)row * C + colo] = cI[j][v] + bv;
                }
            }
        }
    }
}

extern "C" void kernel_launch(void* const* d_in, const int* in_sizes, int n_in,
                              void* d_out, int out_size, void* d_ws, size_t ws_size,
                              hipStream_t stream) {
    const float*     Xr   = (const float*)d_in[0];
    const float*     Xi   = (const float*)d_in[1];
    const float*     Lr   = (const float*)d_in[2];
    const float*     Li   = (const float*)d_in[3];
    const long long* ei   = (const long long*)d_in[4];
    const float*     W    = (const float*)d_in[5];
    const float*     bias = (const float*)d_in[6];
    float*           out  = (float*)d_out;

    const int N = in_sizes[0] / C;
    const int E = in_sizes[4] / 2;

    float* accR = (float*)d_ws;                       // [K1][N][C] f32
    float* accI = accR + (long)K1 * N * C;            // [K1][N][C] f32

    // 1) zero accumulators (153.6 MB, ~7 us at HBM speed)
    long zn = 2L * K1 * N * C;
    MagConv_zero_ws<<<(int)((zn + 255) / 256), 256, 0, stream>>>(accR, zn);

    // 2) fused gather + complex-combine + segment-sum scatter
    long st = (long)E * 32;
    MagConv_scatter<<<(int)((st + 255) / 256), 256, 0, stream>>>(
        Xr, Xi, Lr, Li, ei, accR, accI, N, E);

    // 3) WMMA dual-GEMM tail + bias
    int strips = (N + 15) / 16;
    int blocks = (strips + 7) / 8;
    MagConv_wmma_gemm<<<blocks, 256, 0, stream>>>(
        accR, accI, W, bias, out, out + (long)N * C, N);
}
